// NewMultiHeadAttention_85023172592604
// MI455X (gfx1250) — matrix operand order repaired and checked by reading, not compiled, not run on hardware
//
#include <hip/hip_runtime.h>
#include <hip/hip_bf16.h>
#include <cstdint>

// ---------------------------------------------------------------------------
// MHA forward for MI455X (gfx1250, wave32, WMMA + async/TDM data movers).
//   DIM=768, HEADS=12, HEAD_DIM=64, B=4, T=2048  -> M = B*T = 8192 rows.
//   cvt(fp32->bf16) -> QKV GEMM (V stored transposed) -> flash attn -> proj.
//   Matrix math: v_wmma_f32_16x16x32_bf16 (fp32 accumulate).
//   Staging: tensor_load_to_lds (TDM, TENSORcnt) for attention K/V tiles,
//            global_load_async_to_lds_b128 (ASYNCcnt) for GEMM tiles,
//            double-buffered so the next tile is in flight during compute.
//   Fragments loaded as explicit 16B vector loads per the ISA layout maps.
// ---------------------------------------------------------------------------

typedef __attribute__((ext_vector_type(16))) __bf16 v16bf;
typedef __attribute__((ext_vector_type(8)))  float  v8f;
typedef __bf16 bf16;
typedef unsigned int u32;
typedef u32 u32x4 __attribute__((ext_vector_type(4)));
typedef int  i32x4 __attribute__((ext_vector_type(4)));
typedef int  i32x8 __attribute__((ext_vector_type(8)));

#define DIMC   768
#define NHEADS 12
#define HDIM   64
#define BATCH  4
#define SEQ    2048
#define MROWS  (BATCH * SEQ)   /* 8192 */
#define NQKV   (3 * DIMC)      /* 2304 */

#define HAVE_TDM __has_builtin(__builtin_amdgcn_tensor_load_to_lds)

union FragU {
    uint4 q[2];
    v16bf v;
};

__device__ __forceinline__ v8f wmma_bf16(v16bf a, v16bf b, v8f c) {
    return __builtin_amdgcn_wmma_f32_16x16x32_bf16(
        false, a, false, b, (short)0, c, false, false);
}

// A-frag (16x32, ISA §7.12.2): lane m = lane&15; per lane two contiguous 16B
// chunks at element offsets (m*ld + col0 + grp) and (.. + 16 + grp),
// grp = (lane>>4)*8.  Works for LDS or global pointers (16B-aligned).
__device__ __forceinline__ v16bf load_a_rm(const bf16* s, int row0, int ld,
                                           int col0, int lane) {
    const bf16* p = s + (size_t)(row0 + (lane & 15)) * ld + col0 + ((lane >> 4) << 3);
    FragU u;
    u.q[0] = *(const uint4*)(p);
    u.q[1] = *(const uint4*)(p + 16);
    return u.v;
}

// B-frag (32x16) from transposed storage [n][ld_k] (W[n][k], K[key][d],
// V^T[d][key]): lane n = lane&15; two contiguous 16B chunks at
// (n*ld + col0 + (lane>>4)*8) and (.. + 16).
__device__ __forceinline__ v16bf load_b_tr(const bf16* s, int n0, int ld,
                                           int col0, int lane) {
    const bf16* p = s + (size_t)(n0 + (lane & 15)) * ld + col0 + ((lane >> 4) << 3);
    FragU u;
    u.q[0] = *(const uint4*)(p);
    u.q[1] = *(const uint4*)(p + 16);
    return u.v;
}

// ---- CDNA5 async copy: global -> LDS, 16B per lane, tracked by ASYNCcnt ----
__device__ __forceinline__ void async_ld16(void* lds, const void* g) {
    const u32 l = (u32)(uintptr_t)lds;   // low 32 bits of generic ptr = LDS offset
    asm volatile("global_load_async_to_lds_b128 %0, %1, off"
                 :: "v"(l), "v"(g) : "memory");
}
#define WAIT_ASYNC(n) asm volatile("s_wait_asynccnt %0" :: "n"(n) : "memory")

#if HAVE_TDM
// ---- TDM 2D tile load: tile1 rows of tile0 elements (bf16), row stride s0 --
__device__ __forceinline__ void tdm_load_2d(u32 lds_off, const bf16* gptr,
                                            u32 tile0, u32 tile1, u32 stride0) {
    const unsigned long long ga = (unsigned long long)(uintptr_t)gptr;
    const u32 td0 = stride0;      // >= tile0, tile fully in-bounds
    const u32 td1 = tile1;
    u32x4 g0;
    g0[0] = 1u;                                   // count=1, user-mode D#
    g0[1] = lds_off;                              // lds_addr (bytes)
    g0[2] = (u32)(ga & 0xffffffffu);              // global_addr[31:0]
    g0[3] = (u32)((ga >> 32) & 0x1ffffffu) | (2u << 30);  // addr[56:32] | type=2
    i32x8 g1;
    g1[0] = (int)(1u << 16);                      // wg_mask=0, data_size=1 (2B)
    g1[1] = (int)((td0 & 0xffffu) << 16);         // tensor_dim0[15:0]
    g1[2] = (int)(((td0 >> 16) & 0xffffu) | ((td1 & 0xffffu) << 16));
    g1[3] = (int)(((td1 >> 16) & 0xffffu) | ((tile0 & 0xffffu) << 16));
    g1[4] = (int)(tile1 & 0xffffu);               // tile_dim1 (tile_dim2 = 0)
    g1[5] = (int)stride0;                         // tensor_dim0_stride[31:0]
    g1[6] = 0;
    g1[7] = 0;
    i32x4 g2 = {0, 0, 0, 0};
    i32x4 g3 = {0, 0, 0, 0};
    i32x8 g4 = {0, 0, 0, 0, 0, 0, 0, 0};          // unused VADDR4 group
    __builtin_amdgcn_tensor_load_to_lds(g0, g1, g2, g3, g4, 0);
}
#endif

// ---------------------------------------------------------------------------
__global__ void __launch_bounds__(256)
mha_cvt_bf16(const float* __restrict__ src, bf16* __restrict__ dst, int n) {
    const int i = blockIdx.x * blockDim.x + threadIdx.x;
    if (i < n) dst[i] = (bf16)src[i];
}

// ---------------------------------------------------------------------------
// QKV GEMM: qkv[m, o] = x[m, :] . qkv_w[o, :] + qkv_b[o]
// WG tile 128x64 (4 waves x 32x64), k-step 64, double-buffered async LDS.
// Q/K written [B*H, T, 64]; V written transposed [B*H, 64, T].
__global__ void __launch_bounds__(128)
mha_qkv_gemm(const bf16* __restrict__ X, const bf16* __restrict__ W,
             const float* __restrict__ bias,
             bf16* __restrict__ qb, bf16* __restrict__ kb, bf16* __restrict__ vb) {
    __shared__ bf16 Ash[2][128 * 64];     // 32 KB
    __shared__ bf16 Wsh[2][64 * 64];      // 16 KB
    const int tid  = threadIdx.x;
    const int lane = tid & 31;
    const int wave = tid >> 5;
    const int m0   = blockIdx.y * 128;
    const int n0   = blockIdx.x * 64;

    v8f acc[2][4];
#pragma unroll
    for (int st = 0; st < 2; ++st)
#pragma unroll
        for (int i = 0; i < 4; ++i)
#pragma unroll
            for (int r = 0; r < 8; ++r) acc[st][i][r] = 0.0f;

    auto stage = [&](int buf, int kk) {   // 12 async b128 per wave per tile
#pragma unroll
        for (int i = 0; i < 8; ++i) {     // A: 128x64 tile
            const int f = tid * 8 + i * 1024;
            const int r = f >> 6, c = f & 63;
            async_ld16(&Ash[buf][f], X + (size_t)(m0 + r) * DIMC + kk + c);
        }
#pragma unroll
        for (int i = 0; i < 4; ++i) {     // W: 64x64 tile
            const int f = tid * 8 + i * 1024;
            const int r = f >> 6, c = f & 63;
            async_ld16(&Wsh[buf][f], W + (size_t)(n0 + r) * DIMC + kk + c);
        }
    };

    stage(0, 0);
    int buf = 0;
    for (int kk = 0; kk < DIMC; kk += 64) {
        const bool more = (kk + 64 < DIMC);
        if (more) { stage(buf ^ 1, kk + 64); WAIT_ASYNC(12); }  // next tile in flight
        else      { WAIT_ASYNC(0); }
        __syncthreads();
#pragma unroll
        for (int kc = 0; kc < 64; kc += 32) {
            const v16bf a0 = load_a_rm(&Ash[buf][0], wave * 32,      64, kc, lane);
            const v16bf a1 = load_a_rm(&Ash[buf][0], wave * 32 + 16, 64, kc, lane);
#pragma unroll
            for (int nt = 0; nt < 4; ++nt) {
                const v16bf b = load_b_tr(&Wsh[buf][0], nt * 16, 64, kc, lane);
                acc[0][nt] = wmma_bf16(a0, b, acc[0][nt]);
                acc[1][nt] = wmma_bf16(a1, b, acc[1][nt]);
            }
        }
        __syncthreads();
        buf ^= 1;
    }

    // n-tile -> (which, head): 2304 = 3 * 12 * 64; each 64-wide tile = one head
    const int which = blockIdx.x / NHEADS;
    const int h     = blockIdx.x % NHEADS;
#pragma unroll
    for (int st = 0; st < 2; ++st)
#pragma unroll
        for (int nt = 0; nt < 4; ++nt) {
            const int d  = nt * 16 + (lane & 15);
            const float bs = bias[n0 + d];
#pragma unroll
            for (int r = 0; r < 8; ++r) {
                const int mg = m0 + wave * 32 + st * 16 + r + ((lane >> 4) << 3);
                const int bi = mg >> 11;                 // / SEQ
                const int t  = mg & (SEQ - 1);
                const int bh = bi * NHEADS + h;
                const bf16 val = (bf16)(acc[st][nt][r] + bs);
                if (which == 0)      qb[((size_t)bh * SEQ + t) * HDIM + d] = val;
                else if (which == 1) kb[((size_t)bh * SEQ + t) * HDIM + d] = val;
                else                 vb[((size_t)bh * HDIM + d) * SEQ + t] = val;
            }
        }
}

// ---------------------------------------------------------------------------
// Flash attention: one WG per (bh, 64-query block); 4 waves, each 16 q rows.
// K/V^T tiles staged by TDM (double-buffered); scores scaled by 8 (ref bug).
__global__ void __launch_bounds__(128)
mha_attn(const bf16* __restrict__ qb, const bf16* __restrict__ kb,
         const bf16* __restrict__ vb, bf16* __restrict__ ob) {
    __shared__ bf16 Ksh[2][64 * 64];      // [key][d]
    __shared__ bf16 Vsh[2][64 * 64];      // [d][key]  (V^T tile)
    __shared__ bf16 Psh[4][16 * 64];

    const int tid  = threadIdx.x;
    const int lane = tid & 31;
    const int wave = tid >> 5;
    const int bh   = blockIdx.y;
    const int q0   = blockIdx.x * 64;
    const int bi   = bh / NHEADS;
    const int h    = bh % NHEADS;
    const bf16* kbh = kb + (size_t)bh * SEQ * HDIM;   // [T, 64]
    const bf16* vbh = vb + (size_t)bh * HDIM * SEQ;   // [64, T]

    auto stage_kv = [&](int buf, int j0) {
#if HAVE_TDM
        if (wave == 0) {   // TDM: EXEC-ignored, one wave issues for the WG
            tdm_load_2d((u32)(uintptr_t)&Ksh[buf][0], kbh + (size_t)j0 * HDIM,
                        64, 64, 64);                  // contiguous rows
            tdm_load_2d((u32)(uintptr_t)&Vsh[buf][0], vbh + j0,
                        64, 64, SEQ);                 // strided V^T rows
        }
#else
#pragma unroll
        for (int i = 0; i < 4; ++i) {
            const int f = tid * 8 + i * 1024;
            async_ld16(&Ksh[buf][f], kbh + (size_t)j0 * HDIM + f);
            async_ld16(&Vsh[buf][f], vbh + (size_t)(f >> 6) * SEQ + j0 + (f & 63));
        }
#endif
    };
    auto wait_kv = [&](bool more) {
#if HAVE_TDM
        if (wave == 0) {
            if (more) __builtin_amdgcn_s_wait_tensorcnt((short)2);
            else      __builtin_amdgcn_s_wait_tensorcnt((short)0);
        }
#else
        if (more) { WAIT_ASYNC(8); } else { WAIT_ASYNC(0); }
#endif
        __syncthreads();   // publish LDS to all waves
    };

    // Q fragments for this wave's 16 rows (two 32-wide chunks over head dim)
    v16bf qa[2];
    {
        const bf16* qbase = qb + (size_t)bh * SEQ * HDIM
                               + (size_t)(q0 + wave * 16) * HDIM;
#pragma unroll
        for (int c = 0; c < 2; ++c)
            qa[c] = load_a_rm(qbase, 0, HDIM, c * 32, lane);
    }

    float runmax[8], runsum[8];
    v8f outacc[4];
#pragma unroll
    for (int r = 0; r < 8; ++r) { runmax[r] = -1e30f; runsum[r] = 0.0f; }
#pragma unroll
    for (int dt = 0; dt < 4; ++dt)
#pragma unroll
        for (int r = 0; r < 8; ++r) outacc[dt][r] = 0.0f;

    stage_kv(0, 0);
    int buf = 0;
    for (int j0 = 0; j0 < SEQ; j0 += 64) {
        const bool more = (j0 + 64 < SEQ);
        if (more) stage_kv(buf ^ 1, j0 + 64);
        wait_kv(more);

        // S = Q K^T for 4 key sub-tiles of 16
        v8f s[4];
#pragma unroll
        for (int nt = 0; nt < 4; ++nt) {
#pragma unroll
            for (int r = 0; r < 8; ++r) s[nt][r] = 0.0f;
#pragma unroll
            for (int c = 0; c < 2; ++c) {
                const v16bf bk = load_b_tr(&Ksh[buf][0], nt * 16, 64, c * 32, lane);
                s[nt] = wmma_bf16(qa[c], bk, s[nt]);
            }
#pragma unroll
            for (int r = 0; r < 8; ++r) s[nt][r] *= 8.0f;   // * sqrt(HEAD_DIM)
        }

        // online softmax (lanes 0-15 own rows r, lanes 16-31 own rows r+8)
        float corr[8];
#pragma unroll
        for (int r = 0; r < 8; ++r) {
            float tmax = fmaxf(fmaxf(s[0][r], s[1][r]), fmaxf(s[2][r], s[3][r]));
#pragma unroll
            for (int off = 8; off > 0; off >>= 1)
                tmax = fmaxf(tmax, __shfl_xor(tmax, off, 16));
            const float nm = fmaxf(runmax[r], tmax);
            corr[r]   = __expf(runmax[r] - nm);
            runmax[r] = nm;
            float psum = 0.0f;
#pragma unroll
            for (int nt = 0; nt < 4; ++nt) {
                const float p = __expf(s[nt][r] - nm);
                s[nt][r] = p;
                psum += p;
            }
#pragma unroll
            for (int off = 8; off > 0; off >>= 1)
                psum += __shfl_xor(psum, off, 16);
            runsum[r] = runsum[r] * corr[r] + psum;
        }

#pragma unroll
        for (int dt = 0; dt < 4; ++dt)
#pragma unroll
            for (int r = 0; r < 8; ++r) outacc[dt][r] *= corr[r];

        // P: C-layout -> A-layout via per-wave LDS round-trip (bf16)
        bf16* P = &Psh[wave][0];
#pragma unroll
        for (int nt = 0; nt < 4; ++nt)
#pragma unroll
            for (int r = 0; r < 8; ++r) {
                const int m = r + ((lane >> 4) << 3);
                P[m * 64 + nt * 16 + (lane & 15)] = (bf16)s[nt][r];
            }
        asm volatile("s_wait_dscnt 0x0" ::: "memory");   // wave-local LDS RAW

        // out += P @ V   (V^T in LDS -> contiguous B-frags)
#pragma unroll
        for (int c = 0; c < 2; ++c) {
            const v16bf pa = load_a_rm(P, 0, 64, c * 32, lane);
#pragma unroll
            for (int dt = 0; dt < 4; ++dt) {
                const v16bf bv = load_b_tr(&Vsh[buf][0], dt * 16, 64, c * 32, lane);
                outacc[dt] = wmma_bf16(pa, bv, outacc[dt]);
            }
        }
        __syncthreads();   // done reading buf before it is overwritten
        buf ^= 1;
    }

    // normalize + write attn output as [B*T, 768] bf16
#pragma unroll
    for (int dt = 0; dt < 4; ++dt) {
        const int d = dt * 16 + (lane & 15);
#pragma unroll
        for (int r = 0; r < 8; ++r) {
            const int t = q0 + wave * 16 + r + ((lane >> 4) << 3);
            const float o = outacc[dt][r] / runsum[r];
            ob[((size_t)bi * SEQ + t) * DIMC + h * HDIM + d] = (bf16)o;
        }
    }
}

// ---------------------------------------------------------------------------
// Projection GEMM: out[m, n] = attn[m, :] . proj_w[n, :] + proj_b[n]  (fp32)
// Same 128x64 WG / 32x64 wave tiling as the QKV GEMM.
__global__ void __launch_bounds__(128)
mha_proj_gemm(const bf16* __restrict__ A, const bf16* __restrict__ W,
              const float* __restrict__ bias, float* __restrict__ out) {
    __shared__ bf16 Ash[2][128 * 64];
    __shared__ bf16 Wsh[2][64 * 64];
    const int tid  = threadIdx.x;
    const int lane = tid & 31;
    const int wave = tid >> 5;
    const int m0   = blockIdx.y * 128;
    const int n0   = blockIdx.x * 64;

    v8f acc[2][4];
#pragma unroll
    for (int st = 0; st < 2; ++st)
#pragma unroll
        for (int i = 0; i < 4; ++i)
#pragma unroll
            for (int r = 0; r < 8; ++r) acc[st][i][r] = 0.0f;

    auto stage = [&](int buf, int kk) {
#pragma unroll
        for (int i = 0; i < 8; ++i) {
            const int f = tid * 8 + i * 1024;
            const int r = f >> 6, c = f & 63;
            async_ld16(&Ash[buf][f], A + (size_t)(m0 + r) * DIMC + kk + c);
        }
#pragma unroll
        for (int i = 0; i < 4; ++i) {
            const int f = tid * 8 + i * 1024;
            const int r = f >> 6, c = f & 63;
            async_ld16(&Wsh[buf][f], W + (size_t)(n0 + r) * DIMC + kk + c);
        }
    };

    stage(0, 0);
    int buf = 0;
    for (int kk = 0; kk < DIMC; kk += 64) {
        const bool more = (kk + 64 < DIMC);
        if (more) { stage(buf ^ 1, kk + 64); WAIT_ASYNC(12); }
        else      { WAIT_ASYNC(0); }
        __syncthreads();
#pragma unroll
        for (int kc = 0; kc < 64; kc += 32) {
            const v16bf a0 = load_a_rm(&Ash[buf][0], wave * 32,      64, kc, lane);
            const v16bf a1 = load_a_rm(&Ash[buf][0], wave * 32 + 16, 64, kc, lane);
#pragma unroll
            for (int nt = 0; nt < 4; ++nt) {
                const v16bf b = load_b_tr(&Wsh[buf][0], nt * 16, 64, kc, lane);
                acc[0][nt] = wmma_bf16(a0, b, acc[0][nt]);
                acc[1][nt] = wmma_bf16(a1, b, acc[1][nt]);
            }
        }
        __syncthreads();
        buf ^= 1;
    }

#pragma unroll
    for (int st = 0; st < 2; ++st)
#pragma unroll
        for (int nt = 0; nt < 4; ++nt) {
            const int ng = n0 + nt * 16 + (lane & 15);
            const float bs = bias[ng];
#pragma unroll
            for (int r = 0; r < 8; ++r) {
                const int mg = m0 + wave * 32 + st * 16 + r + ((lane >> 4) << 3);
                out[(size_t)mg * DIMC + ng] = acc[st][nt][r] + bs;
            }
        }
}

// ---------------------------------------------------------------------------
extern "C" void kernel_launch(void* const* d_in, const int* in_sizes, int n_in,
                              void* d_out, int out_size, void* d_ws, size_t ws_size,
                              hipStream_t stream) {
    const float* x      = (const float*)d_in[0];
    const float* qkv_w  = (const float*)d_in[1];
    const float* qkv_b  = (const float*)d_in[2];
    const float* proj_w = (const float*)d_in[3];
    const float* proj_b = (const float*)d_in[4];
    float* out = (float*)d_out;

    const size_t NX  = (size_t)MROWS * DIMC;   // 6291456
    const size_t NWQ = (size_t)NQKV * DIMC;    // 1769472
    const size_t NWP = (size_t)DIMC * DIMC;    // 589824

    char* w = (char*)d_ws;
    bf16* xbf   = (bf16*)(w);
    bf16* wqkv  = (bf16*)(w + 2 * (NX));
    bf16* wproj = (bf16*)(w + 2 * (NX + NWQ));
    bf16* qbuf  = (bf16*)(w + 2 * (NX + NWQ + NWP));
    bf16* kbuf  = (bf16*)(w + 2 * (NX + NWQ + NWP + NX));
    bf16* vbuf  = (bf16*)(w + 2 * (NX + NWQ + NWP + 2 * NX));  // [B*H, 64, T]
    bf16* obuf  = (bf16*)(w + 2 * (NX + NWQ + NWP + 3 * NX));
    (void)in_sizes; (void)n_in; (void)out_size; (void)ws_size;

    mha_cvt_bf16<<<(int)((NX  + 255) / 256), 256, 0, stream>>>(x, xbf, (int)NX);
    mha_cvt_bf16<<<(int)((NWQ + 255) / 256), 256, 0, stream>>>(qkv_w, wqkv, (int)NWQ);
    mha_cvt_bf16<<<(int)((NWP + 255) / 256), 256, 0, stream>>>(proj_w, wproj, (int)NWP);

    mha_qkv_gemm<<<dim3(NQKV / 64, MROWS / 128), 128, 0, stream>>>(
        xbf, wqkv, qkv_b, qbuf, kbuf, vbuf);

    mha_attn<<<dim3(SEQ / 64, BATCH * NHEADS), 128, 0, stream>>>(
        qbuf, kbuf, vbuf, obuf);

    mha_proj_gemm<<<dim3(DIMC / 64, MROWS / 128), 128, 0, stream>>>(
        obuf, wproj, proj_b, out);
}
